// GCN_28544352649657
// MI455X (gfx1250) — compile-verified, matrix operand order checked by reading, not run on hardware
//
#include <hip/hip_runtime.h>
#include <hip/hip_bf16.h>
#include <math.h>

// CDNA5 / gfx1250: wave32, WMMA 16x16x32 bf16 (codegen-confirmed builtin).
typedef __attribute__((ext_vector_type(16))) __bf16 v16bf;
typedef __attribute__((ext_vector_type(8)))  __bf16 v8bf;
typedef __attribute__((ext_vector_type(8)))  float  v8f;
typedef __attribute__((ext_vector_type(8)))  float  v8ff;

// B-fragment view: 16 bf16 = 2 uint4 (for ds_load_b128 pairs).
union Frag16 {
    v16bf v;
    uint4 q[2];
};

// f32x8 -> bf16x8 via vector convert (lowers to v_cvt_pk_bf16_f32 chain,
// results land in consecutive half-registers -- no v_perm fixups).
__device__ __forceinline__ v8bf cvt8(v8ff f) {
    return __builtin_convertvector(f, v8bf);
}

__device__ __forceinline__ v8ff relu8(v8ff f) {
#pragma unroll
    for (int e = 0; e < 8; ++e) f[e] = fmaxf(f[e], 0.0f);
    return f;
}

__device__ __forceinline__ v16bf cat16(v8bf lo, v8bf hi) {
    return __builtin_shufflevector(lo, hi, 0, 1, 2, 3, 4, 5, 6, 7,
                                           8, 9, 10, 11, 12, 13, 14, 15);
}

// ---------------------------------------------------------------------------
// Weight pre-pack: store W in WMMA B-fragment order so each lane's 16 bf16
// elements are contiguous (two ds_load_b128 per fragment in the GEMM).
// Fragment id f = ct*4 + kt; within fragment: lane (0..31) x elem (0..15).
// elem e of lane: col n = ct*16 + (lane&15), koff = (lane>>4)*8,
//                 k = kt*32 + koff + (e < 8 ? e : 8 + e).
// ---------------------------------------------------------------------------

__global__ void pack_w1(const float* __restrict__ W1, __bf16* __restrict__ W1f) {
    int idx = blockIdx.x * 256 + threadIdx.x;   // 8*4*32*16 = 16384
    int e    = idx & 15;
    int lane = (idx >> 4) & 31;
    int kt   = (idx >> 9) & 3;
    int ct   = idx >> 11;
    int m = lane & 15, hi = lane >> 4;
    int k = kt * 32 + hi * 8 + (e < 8 ? e : 8 + e);
    int n = ct * 16 + m;
    W1f[idx] = (__bf16)W1[k * 128 + n];
}

__global__ void pack_w2(const float* __restrict__ W2, __bf16* __restrict__ W2f) {
    int idx = blockIdx.x * 256 + threadIdx.x;   // 3*4*32*16 = 6144
    int e    = idx & 15;
    int lane = (idx >> 4) & 31;
    int kt   = (idx >> 9) & 3;
    int ct   = idx >> 11;                       // 0..2
    int m = lane & 15, hi = lane >> 4;
    int k = kt * 32 + hi * 8 + (e < 8 ? e : 8 + e);
    int n = ct * 16 + m;
    W2f[idx] = (n < 40) ? (__bf16)W2[k * 40 + n] : (__bf16)0.0f;
}

// ---------------------------------------------------------------------------
// GEMM1: H0 = X[N,128] @ W1[128,128].  Block = 8 waves; block owns a 16-row
// strip, wave w owns cols 16w..16w+15.  A rows clamped (no predication);
// B fragments read from LDS in packed order.
// ---------------------------------------------------------------------------

__global__ void __launch_bounds__(256) gemm1_wmma(const float* __restrict__ X,
                                                  const __bf16* __restrict__ W1f,
                                                  float* __restrict__ H0, int N) {
    __shared__ __bf16 Bs[128 * 128];            // 32 KB, fragment-packed
    {
        const uint4* src = (const uint4*)W1f;
        uint4* dst = (uint4*)Bs;
#pragma unroll
        for (int i = 0; i < 8; ++i) dst[threadIdx.x + i * 256] = src[threadIdx.x + i * 256];
    }
    __syncthreads();

    const int wave = threadIdx.x >> 5;
    const int lane = threadIdx.x & 31;
    const int m    = lane & 15;
    const int hi   = lane >> 4;
    const int rowBase = blockIdx.x * 16;
    const int n  = wave * 16 + m;

    int arow = rowBase + m;
    if (arow > N - 1) arow = N - 1;             // clamp: junk rows masked at store
    const float* ap = X + (size_t)arow * 128 + hi * 8;

    v8f c = {};
#pragma unroll
    for (int kt = 0; kt < 4; ++kt) {
        v8ff f0 = *(const v8ff*)(ap + kt * 32);       // K = kt*32+koff ..+7
        v8ff f1 = *(const v8ff*)(ap + kt * 32 + 16);  // K = kt*32+16+koff ..+7
        v16bf a = cat16(cvt8(f0), cvt8(f1));

        Frag16 b;
        const __bf16* bp = Bs + (size_t)((wave * 4 + kt) * 32 + lane) * 16;
        b.q[0] = *(const uint4*)(bp);
        b.q[1] = *(const uint4*)(bp + 8);

        c = __builtin_amdgcn_wmma_f32_16x16x32_bf16(false, a, false, b.v,
                                                    (short)0, c, false, false);
    }

    if (rowBase + 16 <= N) {
#pragma unroll
        for (int r = 0; r < 8; ++r)
            H0[(size_t)(rowBase + r + hi * 8) * 128 + n] = c[r];
    } else {
#pragma unroll
        for (int r = 0; r < 8; ++r) {
            int row = rowBase + r + hi * 8;
            if (row < N) H0[(size_t)row * 128 + n] = c[r];
        }
    }
}

// ---------------------------------------------------------------------------
// Bias broadcast inits (accumulators start at bias).
// ---------------------------------------------------------------------------

__global__ void init_bias128(float* __restrict__ H1, const float* __restrict__ b1, size_t total) {
    size_t i = (size_t)blockIdx.x * 256 + threadIdx.x;
    if (i < total) H1[i] = b1[i & 127];
}

__global__ void init_bias40(float* __restrict__ A2, const float* __restrict__ b2, size_t total) {
    size_t i = (size_t)blockIdx.x * 256 + threadIdx.x;
    if (i < total) A2[i] = b2[i % 40];
}

// ---------------------------------------------------------------------------
// SPMM1: one wave per edge, lane -> 4 dims (b128 gather + 4x atomic fadd).
// ---------------------------------------------------------------------------

__global__ void __launch_bounds__(256) spmm1_scatter(const float* __restrict__ H0,
                                                     const int* __restrict__ ei,
                                                     const float* __restrict__ ew,
                                                     float* __restrict__ H1, int E) {
    long long gw = (long long)blockIdx.x * 8 + (threadIdx.x >> 5);
    int lane = threadIdx.x & 31;
    if (gw >= E) return;
    int r  = ei[gw];
    int cI = ei[(size_t)E + gw];
    float w = ew[gw];
    const float4 v = *(const float4*)(H0 + (size_t)cI * 128 + lane * 4);
    float* dst = H1 + (size_t)r * 128 + lane * 4;
    unsafeAtomicAdd(dst + 0, v.x * w);              // global_atomic_add_f32
    unsafeAtomicAdd(dst + 1, v.y * w);
    unsafeAtomicAdd(dst + 2, v.z * w);
    unsafeAtomicAdd(dst + 3, v.w * w);
}

// ---------------------------------------------------------------------------
// GEMM2: H2 = relu(H1)[N,128] @ W2[128,40] (ReLU fused into bf16 convert).
// One wave per (row-tile, col-tile); 3 col tiles cover 48 padded cols.
// ---------------------------------------------------------------------------

__global__ void __launch_bounds__(256) gemm2_wmma(const float* __restrict__ H1,
                                                  const __bf16* __restrict__ W2f,
                                                  float* __restrict__ H2, int N, int numTiles) {
    __shared__ __bf16 Bs[3 * 4 * 32 * 16];          // 12 KB, fragment-packed
    {
        const uint4* src = (const uint4*)W2f;
        uint4* dst = (uint4*)Bs;
#pragma unroll
        for (int i = 0; i < 3; ++i) dst[threadIdx.x + i * 256] = src[threadIdx.x + i * 256];
    }
    __syncthreads();

    const int wave = threadIdx.x >> 5;
    const int lane = threadIdx.x & 31;
    const int wg = blockIdx.x * 8 + wave;
    if (wg >= numTiles) return;
    const int rt = wg / 3;
    const int ct = wg - rt * 3;

    const int m    = lane & 15;
    const int hi   = lane >> 4;
    const int rowBase = rt * 16;
    const int n  = ct * 16 + m;                     // padded col < 48

    int arow = rowBase + m;
    if (arow > N - 1) arow = N - 1;
    const float* ap = H1 + (size_t)arow * 128 + hi * 8;

    v8f c = {};
#pragma unroll
    for (int kt = 0; kt < 4; ++kt) {
        v8ff f0 = relu8(*(const v8ff*)(ap + kt * 32));
        v8ff f1 = relu8(*(const v8ff*)(ap + kt * 32 + 16));
        v16bf a = cat16(cvt8(f0), cvt8(f1));

        Frag16 b;
        const __bf16* bp = Bs + (size_t)((ct * 4 + kt) * 32 + lane) * 16;
        b.q[0] = *(const uint4*)(bp);
        b.q[1] = *(const uint4*)(bp + 8);

        c = __builtin_amdgcn_wmma_f32_16x16x32_bf16(false, a, false, b.v,
                                                    (short)0, c, false, false);
    }

    if (n < 40) {
#pragma unroll
        for (int r = 0; r < 8; ++r) {
            int row = rowBase + r + hi * 8;
            if (row < N) H2[(size_t)row * 40 + n] = c[r];
        }
    }
}

// ---------------------------------------------------------------------------
// SPMM2: one wave per edge, 40 dims = lane + (lane<8) extra.
// ---------------------------------------------------------------------------

__global__ void __launch_bounds__(256) spmm2_scatter(const float* __restrict__ H2,
                                                     const int* __restrict__ ei,
                                                     const float* __restrict__ ew,
                                                     float* __restrict__ A2, int E) {
    long long gw = (long long)blockIdx.x * 8 + (threadIdx.x >> 5);
    int lane = threadIdx.x & 31;
    if (gw >= E) return;
    int r  = ei[gw];
    int cI = ei[(size_t)E + gw];
    float w = ew[gw];
    const float* src = H2 + (size_t)cI * 40;
    float* dst = A2 + (size_t)r * 40;
    unsafeAtomicAdd(dst + lane, src[lane] * w);
    if (lane < 8) unsafeAtomicAdd(dst + 32 + lane, src[32 + lane] * w);
}

// ---------------------------------------------------------------------------
// log_softmax over 40 cols: one wave per row, wave32 shfl reductions.
// ---------------------------------------------------------------------------

__global__ void __launch_bounds__(256) logsoftmax40(const float* __restrict__ A2,
                                                    float* __restrict__ out, int N) {
    int gw = blockIdx.x * 8 + (threadIdx.x >> 5);
    int lane = threadIdx.x & 31;
    if (gw >= N) return;
    const float* z = A2 + (size_t)gw * 40;
    float z0 = z[lane];
    float z1 = (lane < 8) ? z[32 + lane] : -INFINITY;
    float mx = fmaxf(z0, z1);
#pragma unroll
    for (int o = 16; o; o >>= 1) mx = fmaxf(mx, __shfl_xor(mx, o, 32));
    float s = __expf(z0 - mx) + ((lane < 8) ? __expf(z1 - mx) : 0.0f);
#pragma unroll
    for (int o = 16; o; o >>= 1) s += __shfl_xor(s, o, 32);
    float lg = mx + __logf(s);
    out[(size_t)gw * 40 + lane] = z0 - lg;
    if (lane < 8) out[(size_t)gw * 40 + 32 + lane] = z1 - lg;
}

// ---------------------------------------------------------------------------
// Host-side orchestration.
// ---------------------------------------------------------------------------

extern "C" void kernel_launch(void* const* d_in, const int* in_sizes, int n_in,
                              void* d_out, int out_size, void* d_ws, size_t ws_size,
                              hipStream_t stream) {
    const float* x  = (const float*)d_in[0];
    const int*   ei = (const int*)d_in[1];   // [2,E] row-major: rows then cols
    const float* ew = (const float*)d_in[2];
    const float* W1 = (const float*)d_in[3];
    const float* b1 = (const float*)d_in[4];
    const float* W2 = (const float*)d_in[5];
    const float* b2 = (const float*)d_in[6];
    float* out = (float*)d_out;

    const int N = in_sizes[0] / 128;
    const int E = in_sizes[2];

    // workspace: H0[N*128] | H1[N*128] | H2[N*40] | A2[N*40] | packed bf16 weights
    char* ws = (char*)d_ws;
    float* H0 = (float*)ws;
    float* H1 = H0 + (size_t)N * 128;
    float* H2 = H1 + (size_t)N * 128;
    float* A2 = H2 + (size_t)N * 40;
    size_t off = (size_t)((char*)(A2 + (size_t)N * 40) - ws);
    off = (off + 31) & ~(size_t)31;          // 32B align for fragment vectors
    __bf16* W1f = (__bf16*)(ws + off);
    __bf16* W2f = W1f + 128 * 128;

    dim3 blk(256);
    const int rowTiles = (N + 15) / 16;

    pack_w1<<<64, blk, 0, stream>>>(W1, W1f);
    pack_w2<<<24, blk, 0, stream>>>(W2, W2f);

    gemm1_wmma<<<rowTiles, blk, 0, stream>>>(x, W1f, H0, N);

    const size_t t1 = (size_t)N * 128;
    init_bias128<<<(unsigned)((t1 + 255) / 256), blk, 0, stream>>>(H1, b1, t1);
    spmm1_scatter<<<(E + 7) / 8, blk, 0, stream>>>(H0, ei, ew, H1, E);

    const int tiles2 = rowTiles * 3;
    gemm2_wmma<<<(tiles2 + 7) / 8, blk, 0, stream>>>(H1, W2f, H2, N, tiles2);

    const size_t t2 = (size_t)N * 40;
    init_bias40<<<(unsigned)((t2 + 255) / 256), blk, 0, stream>>>(A2, b2, t2);
    spmm2_scatter<<<(E + 7) / 8, blk, 0, stream>>>(H2, ei, ew, A2, E);

    logsoftmax40<<<(N + 7) / 8, blk, 0, stream>>>(A2, out, N);
}
